// EquivariantPartitionNet_29566554866420
// MI455X (gfx1250) — matrix operand order repaired and checked by reading, not compiled, not run on hardware
//
#include <hip/hip_runtime.h>
#include <hip/hip_bf16.h>

typedef _Float16 half_t;
typedef __attribute__((ext_vector_type(16))) _Float16 v16h;
typedef __attribute__((ext_vector_type(8)))  float    v8f;
typedef __attribute__((ext_vector_type(8)))  unsigned int v8u;

#define NB 2
#define NN 320
#define JT (NN / 16)
#define EDGE_TILES (NB * NN * JT)

// ---------------- WMMA fragment helpers (gfx1250, wave32) ----------------
// A: 16x32 f16 tile from LDS, row-major with given stride.
// contiguous k order, same as fragB: lanes 0-15 -> M=lane, K 0..15; lanes 16-31 -> M=lane-16, K 16..31
__device__ __forceinline__ v16h fragA(const half_t* sbase, int stride, int kb) {
  int lane = threadIdx.x & 31;
  const half_t* p = sbase + (lane & 15) * stride + kb + ((lane >> 4) << 4);
  const unsigned int* q0 = (const unsigned int*)p;
  const unsigned int* q1 = (const unsigned int*)(p + 8);
  v8u u;
  u[0] = q0[0]; u[1] = q0[1]; u[2] = q0[2]; u[3] = q0[3];
  u[4] = q1[0]; u[5] = q1[1]; u[6] = q1[2]; u[7] = q1[3];
  return __builtin_bit_cast(v16h, u);
}

// B: 32x16 f16 tile from global weights stored transposed as WT[n][Kpad].
// lanes 0-15 -> N=lane, K=kb..kb+15 ; lanes 16-31 -> N=lane-16, K=kb+16..kb+31
__device__ __forceinline__ v16h fragB(const half_t* wt, int Kpad, int nb, int kb) {
  int lane = threadIdx.x & 31;
  const unsigned int* q =
      (const unsigned int*)(wt + (size_t)(nb + (lane & 15)) * Kpad + kb + ((lane >> 4) << 4));
  v8u u;
#pragma unroll
  for (int r = 0; r < 8; ++r) u[r] = q[r];
  return __builtin_bit_cast(v16h, u);
}

__device__ __forceinline__ v8f wmma_f16(v16h a, v16h b, v8f c) {
  return __builtin_amdgcn_wmma_f32_16x16x32_f16(false, a, false, b, (short)0, c, false, false);
}

// Dual-tile GEMM: one A-fragment load feeds two WMMAs (cols nb0 and nb0+16).
__device__ __forceinline__ void gemm_2tile(const half_t* A, int stride,
                                           const half_t* WT, int Kpad, int nb0,
                                           int ktiles, v8f& acc0, v8f& acc1) {
  for (int kt = 0; kt < ktiles; ++kt) {
    v16h a = fragA(A, stride, kt * 32);
    acc0 = wmma_f16(a, fragB(WT, Kpad, nb0, kt * 32), acc0);
    acc1 = wmma_f16(a, fragB(WT, Kpad, nb0 + 16, kt * 32), acc1);
  }
}

__device__ __forceinline__ float silu_f(float v) { return v / (1.f + __expf(-v)); }

// Row-wise LayerNorm over 256 cols of fp32 LDS tile [16][256]; 16 threads/row,
// write f16 result to outH[16][256].
__device__ __forceinline__ void ln256_to_f16(const float* sF, half_t* outH,
                                             const float* g, const float* bta) {
  int tid = threadIdx.x;
  int rrow = tid >> 4, s = tid & 15;
  const float* rp = sF + rrow * 256 + s * 16;
  float sum = 0.f;
#pragma unroll
  for (int q = 0; q < 16; ++q) sum += rp[q];
#pragma unroll
  for (int mk = 1; mk < 16; mk <<= 1) sum += __shfl_xor(sum, mk, 32);
  float mean = sum * (1.f / 256.f);
  float ss = 0.f;
#pragma unroll
  for (int q = 0; q < 16; ++q) { float d = rp[q] - mean; ss += d * d; }
#pragma unroll
  for (int mk = 1; mk < 16; mk <<= 1) ss += __shfl_xor(ss, mk, 32);
  float rstd = rsqrtf(ss * (1.f / 256.f) + 1e-5f);
#pragma unroll
  for (int q = 0; q < 16; ++q) {
    int c = s * 16 + q;
    outH[rrow * 256 + c] = (half_t)((rp[q] - mean) * rstd * g[c] + bta[c]);
  }
}

// ---------------- weight prep: fp32 W[K][Nout] -> f16 WT[NoutPad][Kpad] ----------------
__global__ void prep_wt(const float* __restrict__ W, half_t* __restrict__ WT,
                        int K, int Nout, int NoutPad, int Kpad) {
  int idx = blockIdx.x * 256 + threadIdx.x;
  int total = NoutPad * Kpad;
  if (idx >= total) return;
  int n = idx / Kpad, k = idx % Kpad;
  float v = (k < K && n < Nout) ? W[(size_t)k * Nout + n] : 0.f;
  WT[idx] = (half_t)v;
}

// ---------------- fused per-edge kernel ----------------
__global__ void __launch_bounds__(256)
edge_kernel(const float* __restrict__ h, const float* __restrict__ x, const float* __restrict__ e,
            const half_t* wtM1, const float* mb1, const float* mlg, const float* mlb,
            const half_t* wtM2, const float* mb2,
            const half_t* wtM3, const float* mb3,
            const half_t* wtA1, const float* ab1,
            const half_t* wtA2, const float* ab2,
            const half_t* wtC1, const float* cb1, const float* cw2,
            const half_t* wtE1, const float* eb1, const float* elg, const float* elb,
            const half_t* wtE2, const float* eb2, const float* eng, const float* enb,
            float* __restrict__ msgOut, float* __restrict__ logitOut,
            float* __restrict__ cwOut, float* __restrict__ eNewOut) {
  __shared__ __align__(16) half_t sIn[16 * 544];   // msg_in tile (later reused as coord act [16][128])
  __shared__ __align__(16) half_t sAct[16 * 256];  // post-LN act / messages f16
  __shared__ __align__(16) half_t sAct2[16 * 256]; // act2 / attn act / edge-LN act
  __shared__ __align__(16) half_t sCat[16 * 512];  // [e | messages] concat f16
  __shared__ __align__(16) float  sF32[16 * 256];  // fp32 staging for LayerNorms

  int t = blockIdx.x;
  int b = t / (NN * JT);
  int rem = t % (NN * JT);
  int i = rem / JT;
  int j0 = (rem % JT) * 16;
  int tid = threadIdx.x;
  int wave = tid >> 5;
  int lane = tid & 31;
  int col = lane & 15;
  int row0 = (lane >> 4) * 8;

  const size_t edgeBase = ((size_t)(b * NN + i)) * NN + j0;  // first edge row index

  float xi0 = x[(b * NN + i) * 2 + 0], xi1 = x[(b * NN + i) * 2 + 1];

  // ---- stage 1: build msg_in = [h_i | h_j | e | dist | pad] ; stash e into sCat ----
  for (int idx = tid; idx < 16 * 544; idx += 256) {
    int m = idx / 544, c = idx % 544;
    int j = j0 + m;
    float v;
    if (c < 128) v = h[((size_t)(b * NN + i)) * 128 + c];
    else if (c < 256) v = h[((size_t)(b * NN + j)) * 128 + (c - 128)];
    else if (c < 512) {
      v = e[(edgeBase + m) * 256 + (c - 256)];
      sCat[m * 512 + (c - 256)] = (half_t)v;
    } else if (c == 512) {
      float dx = x[(b * NN + j) * 2 + 0] - xi0;
      float dy = x[(b * NN + j) * 2 + 1] - xi1;
      v = sqrtf(dx * dx + dy * dy + 1e-12f);
    } else v = 0.f;
    sIn[idx] = (half_t)v;
  }
  __syncthreads();

  // ---- message GEMM1: (16x544)@(544x256) + b1, SiLU -> sF32 ----
  {
    int nb0 = wave * 32;
    v8f a0 = {}, a1 = {};
    gemm_2tile(sIn, 544, wtM1, 544, nb0, 17, a0, a1);
#pragma unroll
    for (int r = 0; r < 8; ++r) {
      sF32[(row0 + r) * 256 + nb0 + col]      = silu_f(a0[r] + mb1[nb0 + col]);
      sF32[(row0 + r) * 256 + nb0 + 16 + col] = silu_f(a1[r] + mb1[nb0 + 16 + col]);
    }
  }
  __syncthreads();
  ln256_to_f16(sF32, sAct, mlg, mlb);
  __syncthreads();

  // ---- message GEMM2: 256x256 + b2, SiLU -> sAct2 f16 ----
  {
    int nb0 = wave * 32;
    v8f a0 = {}, a1 = {};
    gemm_2tile(sAct, 256, wtM2, 256, nb0, 8, a0, a1);
#pragma unroll
    for (int r = 0; r < 8; ++r) {
      sAct2[(row0 + r) * 256 + nb0 + col]      = (half_t)silu_f(a0[r] + mb2[nb0 + col]);
      sAct2[(row0 + r) * 256 + nb0 + 16 + col] = (half_t)silu_f(a1[r] + mb2[nb0 + 16 + col]);
    }
  }
  __syncthreads();

  // ---- message GEMM3: 256x256 + b3 -> messages (global fp32, sAct f16, sCat hi half) ----
  {
    int nb0 = wave * 32;
    v8f a0 = {}, a1 = {};
    gemm_2tile(sAct2, 256, wtM3, 256, nb0, 8, a0, a1);
#pragma unroll
    for (int r = 0; r < 8; ++r) {
      int m = row0 + r;
      {
        int n = nb0 + col;
        float v = a0[r] + mb3[n];
        msgOut[(edgeBase + m) * 256 + n] = v;
        half_t hv = (half_t)v;
        sAct[m * 256 + n] = hv;
        sCat[m * 512 + 256 + n] = hv;
      }
      {
        int n = nb0 + 16 + col;
        float v = a1[r] + mb3[n];
        msgOut[(edgeBase + m) * 256 + n] = v;
        half_t hv = (half_t)v;
        sAct[m * 256 + n] = hv;
        sCat[m * 512 + 256 + n] = hv;
      }
    }
  }
  __syncthreads();

  // ---- attention branch: messages @ attn_w1 (256->128) + b, SiLU -> sAct2 [16][128] ----
  {
    int nb = wave * 16;
    v8f acc = {};
    for (int kt = 0; kt < 8; ++kt)
      acc = wmma_f16(fragA(sAct, 256, kt * 32), fragB(wtA1, 256, nb, kt * 32), acc);
#pragma unroll
    for (int r = 0; r < 8; ++r)
      sAct2[(row0 + r) * 128 + nb + col] = (half_t)silu_f(acc[r] + ab1[nb + col]);
  }
  __syncthreads();

  // ---- attn_w2 (128->8, padded to 16): wave 0 only; store scaled logits ----
  if (wave == 0) {
    v8f acc = {};
    for (int kt = 0; kt < 4; ++kt)
      acc = wmma_f16(fragA(sAct2, 128, kt * 32), fragB(wtA2, 128, 0, kt * 32), acc);
    if (col < 8) {
      const float sc = 0.17677669529663688f;  // 1/sqrt(32)
#pragma unroll
      for (int r = 0; r < 8; ++r)
        logitOut[(edgeBase + row0 + r) * 8 + col] = (acc[r] + ab2[col]) * sc;
    }
  }

  // ---- coord branch: messages @ coord_w1 (256->128) + b, SiLU -> sIn reused [16][128] ----
  {
    half_t* sCrd = sIn;
    int nb = wave * 16;
    v8f acc = {};
    for (int kt = 0; kt < 8; ++kt)
      acc = wmma_f16(fragA(sAct, 256, kt * 32), fragB(wtC1, 256, nb, kt * 32), acc);
#pragma unroll
    for (int r = 0; r < 8; ++r)
      sCrd[(row0 + r) * 128 + nb + col] = (half_t)silu_f(acc[r] + cb1[nb + col]);
  }
  __syncthreads();

  // ---- cw = tanh(dot(act,coord_w2)/10): 16 threads per row ----
  {
    int rrow = tid >> 4, s = tid & 15;
    const half_t* rp = sIn + rrow * 128 + s * 8;
    float dot = 0.f;
#pragma unroll
    for (int q = 0; q < 8; ++q) dot += (float)rp[q] * cw2[s * 8 + q];
#pragma unroll
    for (int mk = 1; mk < 16; mk <<= 1) dot += __shfl_xor(dot, mk, 32);
    if (s == 0) cwOut[edgeBase + rrow] = tanhf(dot * 0.1f);
  }

  // ---- edge update GEMM1: [e|msg] (16x512)@(512x256) + b, SiLU -> sF32 ----
  {
    int nb0 = wave * 32;
    v8f a0 = {}, a1 = {};
    gemm_2tile(sCat, 512, wtE1, 512, nb0, 16, a0, a1);
#pragma unroll
    for (int r = 0; r < 8; ++r) {
      sF32[(row0 + r) * 256 + nb0 + col]      = silu_f(a0[r] + eb1[nb0 + col]);
      sF32[(row0 + r) * 256 + nb0 + 16 + col] = silu_f(a1[r] + eb1[nb0 + 16 + col]);
    }
  }
  __syncthreads();
  ln256_to_f16(sF32, sAct2, elg, elb);
  __syncthreads();

  // ---- edge update GEMM2: 256x256 + b2 + residual e -> sF32 ----
  {
    int nb0 = wave * 32;
    v8f a0 = {}, a1 = {};
    gemm_2tile(sAct2, 256, wtE2, 256, nb0, 8, a0, a1);
#pragma unroll
    for (int r = 0; r < 8; ++r) {
      int m = row0 + r;
      int n0 = nb0 + col, n1 = nb0 + 16 + col;
      sF32[m * 256 + n0] = a0[r] + eb2[n0] + e[(edgeBase + m) * 256 + n0];
      sF32[m * 256 + n1] = a1[r] + eb2[n1] + e[(edgeBase + m) * 256 + n1];
    }
  }
  __syncthreads();

  // ---- final edge LayerNorm -> e_new (global fp32, non-temporal: never re-read) ----
  {
    int rrow = tid >> 4, s = tid & 15;
    const float* rp = sF32 + rrow * 256 + s * 16;
    float sum = 0.f;
#pragma unroll
    for (int q = 0; q < 16; ++q) sum += rp[q];
#pragma unroll
    for (int mk = 1; mk < 16; mk <<= 1) sum += __shfl_xor(sum, mk, 32);
    float mean = sum * (1.f / 256.f);
    float ss = 0.f;
#pragma unroll
    for (int q = 0; q < 16; ++q) { float d = rp[q] - mean; ss += d * d; }
#pragma unroll
    for (int mk = 1; mk < 16; mk <<= 1) ss += __shfl_xor(ss, mk, 32);
    float rstd = rsqrtf(ss * (1.f / 256.f) + 1e-5f);
#pragma unroll
    for (int q = 0; q < 16; ++q) {
      int c = s * 16 + q;
      float v = (rp[q] - mean) * rstd * eng[c] + enb[c];
      __builtin_nontemporal_store(v, &eNewOut[(edgeBase + rrow) * 256 + c]);
    }
  }
}

// ---------------- per-node aggregation: softmax, m_att via folded val_w, coord update ----------------
__global__ void __launch_bounds__(256)
agg_kernel(const float* __restrict__ x, const float* __restrict__ logits,
           const float* __restrict__ msg, const float* __restrict__ cw,
           const half_t* __restrict__ wtVal, const float* __restrict__ vb,
           const float* __restrict__ coordScale,
           float* __restrict__ mAttOut, float* __restrict__ xOut) {
  __shared__ float sAttn[8 * NN];
  __shared__ float sS[8 * 256];
  __shared__ float sRed[16];
  int bi = blockIdx.x;
  int b = bi / NN;
  int tid = threadIdx.x, wave = tid >> 5, lane = tid & 31;
  size_t rowBase = (size_t)bi * NN;

  // softmax over neighbors j, one head per wave
  {
    int hh = wave;
    float mx = -1e30f;
    for (int j = lane; j < NN; j += 32) mx = fmaxf(mx, logits[(rowBase + j) * 8 + hh]);
#pragma unroll
    for (int mk = 1; mk < 32; mk <<= 1) mx = fmaxf(mx, __shfl_xor(mx, mk, 32));
    float sum = 0.f;
    for (int j = lane; j < NN; j += 32) {
      float ev = __expf(logits[(rowBase + j) * 8 + hh] - mx);
      sAttn[hh * NN + j] = ev;
      sum += ev;
    }
#pragma unroll
    for (int mk = 1; mk < 32; mk <<= 1) sum += __shfl_xor(sum, mk, 32);
    float inv = 1.f / sum;
    for (int j = lane; j < NN; j += 32) sAttn[hh * NN + j] *= inv;
  }
  __syncthreads();

  // s[h][k] = sum_j attn[h][j] * messages[j][k]   (thread t == k)
  {
    int k = tid;
    float acc[8] = {0, 0, 0, 0, 0, 0, 0, 0};
    for (int j = 0; j < NN; ++j) {
      float mv = msg[(rowBase + j) * 256 + k];
#pragma unroll
      for (int hh = 0; hh < 8; ++hh) acc[hh] += sAttn[hh * NN + j] * mv;
    }
#pragma unroll
    for (int hh = 0; hh < 8; ++hh) sS[hh * 256 + k] = acc[hh];
  }
  __syncthreads();

  // m_att[n] = s[n>>5] . val_w[:,n] + val_b[n]   (val_w folded through softmax)
  {
    int n = tid, hh = n >> 5;
    const half_t* wv = wtVal + (size_t)n * 256;
    const float* sp = sS + hh * 256;
    float dot = 0.f;
    for (int k = 0; k < 256; ++k) dot += sp[k] * (float)wv[k];
    mAttOut[(size_t)bi * 256 + n] = dot + vb[n];
  }

  // coordinate update
  {
    float xi0 = x[bi * 2 + 0], xi1 = x[bi * 2 + 1];
    float ux = 0.f, uy = 0.f;
    for (int j = tid; j < NN; j += 256) {
      float dx = x[(b * NN + j) * 2 + 0] - xi0;
      float dy = x[(b * NN + j) * 2 + 1] - xi1;
      float dist = sqrtf(dx * dx + dy * dy + 1e-12f);
      float w = cw[rowBase + j] / (dist + 1e-8f);
      ux += w * dx;
      uy += w * dy;
    }
#pragma unroll
    for (int mk = 1; mk < 32; mk <<= 1) {
      ux += __shfl_xor(ux, mk, 32);
      uy += __shfl_xor(uy, mk, 32);
    }
    if (lane == 0) { sRed[wave] = ux; sRed[8 + wave] = uy; }
    __syncthreads();
    if (tid == 0) {
      float sx = 0.f, sy = 0.f;
      for (int w = 0; w < 8; ++w) { sx += sRed[w]; sy += sRed[8 + w]; }
      float cs = coordScale[0];
      xOut[bi * 2 + 0] = xi0 + cs * sx;
      xOut[bi * 2 + 1] = xi1 + cs * sy;
    }
  }
}

// ---------------- node update: concat(h, m_att) -> MLP -> residual LN ----------------
__global__ void __launch_bounds__(256)
node_kernel(const float* __restrict__ h, const float* __restrict__ mAtt,
            const half_t* wtN1, const float* nb1, const float* nlg, const float* nlb,
            const half_t* wtN2, const float* nb2,
            const float* nng, const float* nnb,
            float* __restrict__ hOut) {
  __shared__ __align__(16) half_t sA[16 * 384];
  __shared__ __align__(16) half_t sAct[16 * 256];
  __shared__ __align__(16) float  sF[16 * 256];
  int r0 = blockIdx.x * 16;
  int tid = threadIdx.x, wave = tid >> 5, lane = tid & 31;
  int col = lane & 15, row0 = (lane >> 4) * 8;

  for (int idx = tid; idx < 16 * 384; idx += 256) {
    int m = idx / 384, c = idx % 384;
    int bi = r0 + m;
    float v = (c < 128) ? h[(size_t)bi * 128 + c] : mAtt[(size_t)bi * 256 + (c - 128)];
    sA[idx] = (half_t)v;
  }
  __syncthreads();

  {
    int nb0 = wave * 32;
    v8f a0 = {}, a1 = {};
    gemm_2tile(sA, 384, wtN1, 384, nb0, 12, a0, a1);
#pragma unroll
    for (int r = 0; r < 8; ++r) {
      sF[(row0 + r) * 256 + nb0 + col]      = silu_f(a0[r] + nb1[nb0 + col]);
      sF[(row0 + r) * 256 + nb0 + 16 + col] = silu_f(a1[r] + nb1[nb0 + 16 + col]);
    }
  }
  __syncthreads();
  ln256_to_f16(sF, sAct, nlg, nlb);
  __syncthreads();

  {
    int nb = wave * 16;
    v8f acc = {};
    for (int kt = 0; kt < 8; ++kt)
      acc = wmma_f16(fragA(sAct, 256, kt * 32), fragB(wtN2, 256, nb, kt * 32), acc);
#pragma unroll
    for (int r = 0; r < 8; ++r) {
      int m = row0 + r, n = nb + col;
      sF[m * 128 + n] = acc[r] + nb2[n] + h[(size_t)(r0 + m) * 128 + n];
    }
  }
  __syncthreads();

  {
    int rrow = tid >> 4, s = tid & 15;
    const float* rp = sF + rrow * 128 + s * 8;
    float sum = 0.f;
#pragma unroll
    for (int q = 0; q < 8; ++q) sum += rp[q];
#pragma unroll
    for (int mk = 1; mk < 16; mk <<= 1) sum += __shfl_xor(sum, mk, 32);
    float mean = sum * (1.f / 128.f);
    float ss = 0.f;
#pragma unroll
    for (int q = 0; q < 8; ++q) { float d = rp[q] - mean; ss += d * d; }
#pragma unroll
    for (int mk = 1; mk < 16; mk <<= 1) ss += __shfl_xor(ss, mk, 32);
    float rstd = rsqrtf(ss * (1.f / 128.f) + 1e-5f);
#pragma unroll
    for (int q = 0; q < 8; ++q) {
      int c = s * 8 + q;
      float v = (rp[q] - mean) * rstd * nng[c] + nnb[c];
      __builtin_nontemporal_store(v, &hOut[(size_t)(r0 + rrow) * 128 + c]);
    }
  }
}

extern "C" void kernel_launch(void* const* d_in, const int* in_sizes, int n_in,
                              void* d_out, int out_size, void* d_ws, size_t ws_size,
                              hipStream_t stream) {
  const float* h         = (const float*)d_in[0];
  const float* x         = (const float*)d_in[1];
  const float* e         = (const float*)d_in[2];
  const float* msg_w1    = (const float*)d_in[3];
  const float* msg_b1    = (const float*)d_in[4];
  const float* msg_ln_g  = (const float*)d_in[5];
  const float* msg_ln_b  = (const float*)d_in[6];
  const float* msg_w2    = (const float*)d_in[7];
  const float* msg_b2    = (const float*)d_in[8];
  const float* msg_w3    = (const float*)d_in[9];
  const float* msg_b3    = (const float*)d_in[10];
  const float* attn_w1   = (const float*)d_in[11];
  const float* attn_b1   = (const float*)d_in[12];
  const float* attn_w2   = (const float*)d_in[13];
  const float* attn_b2   = (const float*)d_in[14];
  const float* val_w     = (const float*)d_in[15];
  const float* val_b     = (const float*)d_in[16];
  const float* coord_w1  = (const float*)d_in[17];
  const float* coord_b1  = (const float*)d_in[18];
  const float* coord_w2  = (const float*)d_in[19];
  const float* node_w1   = (const float*)d_in[20];
  const float* node_b1   = (const float*)d_in[21];
  const float* node_ln_g = (const float*)d_in[22];
  const float* node_ln_b = (const float*)d_in[23];
  const float* node_w2   = (const float*)d_in[24];
  const float* node_b2   = (const float*)d_in[25];
  const float* edge_w1   = (const float*)d_in[26];
  const float* edge_b1   = (const float*)d_in[27];
  const float* edge_ln_g = (const float*)d_in[28];
  const float* edge_ln_b = (const float*)d_in[29];
  const float* edge_w2   = (const float*)d_in[30];
  const float* edge_b2   = (const float*)d_in[31];
  const float* node_norm_g = (const float*)d_in[32];
  const float* node_norm_b = (const float*)d_in[33];
  const float* edge_norm_g = (const float*)d_in[34];
  const float* edge_norm_b = (const float*)d_in[35];
  const float* coord_scale = (const float*)d_in[36];

  // workspace layout: transposed f16 weights, then fp32 scratch tensors
  half_t* wp = (half_t*)d_ws;
  half_t* wtM1 = wp; wp += 256 * 544;
  half_t* wtM2 = wp; wp += 256 * 256;
  half_t* wtM3 = wp; wp += 256 * 256;
  half_t* wtA1 = wp; wp += 128 * 256;
  half_t* wtA2 = wp; wp += 16 * 128;
  half_t* wtVal = wp; wp += 256 * 256;
  half_t* wtC1 = wp; wp += 128 * 256;
  half_t* wtN1 = wp; wp += 256 * 384;
  half_t* wtN2 = wp; wp += 128 * 256;
  half_t* wtE1 = wp; wp += 256 * 512;
  half_t* wtE2 = wp; wp += 256 * 256;
  size_t off = (((size_t)wp - (size_t)d_ws) + 15) & ~(size_t)15;
  float* msg    = (float*)((char*)d_ws + off);
  float* logits = msg + (size_t)NB * NN * NN * 256;
  float* cw     = logits + (size_t)NB * NN * NN * 8;
  float* mAtt   = cw + (size_t)NB * NN * NN;

  float* hOut = (float*)d_out;
  float* xOut = hOut + (size_t)NB * NN * 128;
  float* eOut = xOut + (size_t)NB * NN * 2;

  auto prep = [&](const float* W, half_t* WT, int K, int Nout, int NoutPad, int Kpad) {
    int total = NoutPad * Kpad;
    prep_wt<<<(total + 255) / 256, 256, 0, stream>>>(W, WT, K, Nout, NoutPad, Kpad);
  };
  prep(msg_w1,  wtM1, 513, 256, 256, 544);
  prep(msg_w2,  wtM2, 256, 256, 256, 256);
  prep(msg_w3,  wtM3, 256, 256, 256, 256);
  prep(attn_w1, wtA1, 256, 128, 128, 256);
  prep(attn_w2, wtA2, 128, 8,   16,  128);
  prep(val_w,   wtVal, 256, 256, 256, 256);
  prep(coord_w1, wtC1, 256, 128, 128, 256);
  prep(node_w1, wtN1, 384, 256, 256, 384);
  prep(node_w2, wtN2, 256, 128, 128, 256);
  prep(edge_w1, wtE1, 512, 256, 256, 512);
  prep(edge_w2, wtE2, 256, 256, 256, 256);

  edge_kernel<<<EDGE_TILES, 256, 0, stream>>>(
      h, x, e,
      wtM1, msg_b1, msg_ln_g, msg_ln_b,
      wtM2, msg_b2, wtM3, msg_b3,
      wtA1, attn_b1, wtA2, attn_b2,
      wtC1, coord_b1, coord_w2,
      wtE1, edge_b1, edge_ln_g, edge_ln_b,
      wtE2, edge_b2, edge_norm_g, edge_norm_b,
      msg, logits, cw, eOut);

  agg_kernel<<<NB * NN, 256, 0, stream>>>(x, logits, msg, cw, wtVal, val_b, coord_scale,
                                          mAtt, xOut);

  node_kernel<<<(NB * NN) / 16, 256, 0, stream>>>(
      h, mAtt, wtN1, node_b1, node_ln_g, node_ln_b,
      wtN2, node_b2, node_norm_g, node_norm_b, hOut);
}
